// VectorQuantizer_17102559772722
// MI455X (gfx1250) — matrix operand order repaired and checked by reading, not compiled, not run on hardware
//
#include <hip/hip_runtime.h>

// VQ codebook lookup for MI455X (gfx1250).
// bf16 split-3 WMMA (hi*hi + hi*lo + lo*hi, f32 accum) for near-f32 accuracy
// on the 103-GFLOP score matmul; ordered-key u64 LDS atomic-min argmin;
// straight-through gather reproducing reference FP ordering.
//
// Round-1 fix: the compiler fully unrolled the K loop and spilled A fragments
// to scratch (scratch_load_b128 before every WMMA). Force unroll(1) on the K
// loop and double-buffer the L2-resident B fragments instead.

#define N_TOKENS 262144
#define EMB_DIM  384
#define NUM_CODES 512
#define BETA 0.25f

#define KCHUNKS 12            // 384 / 32
#define TOK_PER_CTA 128
#define ATILES 8              // 128 / 16

typedef __bf16 v16bf __attribute__((ext_vector_type(16)));
typedef float  v8f   __attribute__((ext_vector_type(8)));

// d_ws layout in 32-bit words:
//   [0, 512)                : e_sq (f32)
//   [512, 512+98304)        : B hi fragments (packed bf16 pairs)
//   [98816, 98816+98304)    : B lo fragments
//   [197120, 197120+2048)   : per-block loss partials (f32)
#define WS_ESQ  0
#define WS_BHI  512
#define WS_BLO  (512 + 98304)
#define WS_PART (512 + 2 * 98304)

__device__ __forceinline__ unsigned bf16_rne(float x) {
  unsigned u = __float_as_uint(x);
  return (u + 0x7FFFu + ((u >> 16) & 1u)) >> 16;
}
__device__ __forceinline__ float bf16_to_f(unsigned b) {
  return __uint_as_float(b << 16);
}
// Split two consecutive f32 into packed bf16 hi-pair and lo-pair words.
__device__ __forceinline__ void split_pack(float a, float b,
                                           unsigned &hi, unsigned &lo) {
  unsigned ah = bf16_rne(a), bh = bf16_rne(b);
  float al = a - bf16_to_f(ah);
  float bl = b - bf16_to_f(bh);
  unsigned alb = bf16_rne(al), blb = bf16_rne(bl);
  hi = (bh << 16) | ah;
  lo = (blb << 16) | alb;
}

// ---------------------------------------------------------------------------
// Kernel 1: build B fragments (emb, bf16 hi/lo, WMMA B 32x16 layout) + e_sq.
// One block per codebook row n; 192 threads = 192 K-pairs.
// B layout (16-bit B, 32x16): lane = 16*(k>=16 within chunk) + n%16,
// word w holds K pair {2w, 2w+1} (plus 16 for the upper lane half).
// ---------------------------------------------------------------------------
__global__ __launch_bounds__(192) void vq_prep(const float* __restrict__ emb,
                                               unsigned* __restrict__ ws) {
  __shared__ float red[192];
  const int n = blockIdx.x;
  const int t = threadIdx.x;           // pair index 0..191
  const int kc = t >> 4;
  const int r  = t & 15;
  const int h  = r >> 3;
  const int w  = r & 7;
  const int k  = kc * 32 + 16 * h + 2 * w;

  const float e0 = emb[n * EMB_DIM + k];
  const float e1 = emb[n * EMB_DIM + k + 1];
  unsigned hiw, low;
  split_pack(e0, e1, hiw, low);

  const int ct   = n >> 4;
  const int lane = 16 * h + (n & 15);
  const int off  = ((ct * KCHUNKS + kc) * 32 + lane) * 8 + w;
  ws[WS_BHI + off] = hiw;
  ws[WS_BLO + off] = low;

  red[t] = e0 * e0 + e1 * e1;
  __syncthreads();
  if (t == 0) {
    float s = 0.f;
    for (int i = 0; i < 192; ++i) s += red[i];   // fixed order: deterministic
    ((float*)ws)[WS_ESQ + n] = s;
  }
}

// ---------------------------------------------------------------------------
// Kernel 2: per-token argmin over 512 codes via bf16 split-3 WMMA.
// CTA: 256 threads (8 waves), 128 tokens. LDS: A frags (192 KB) + u64 argmin.
// A layout (16-bit A, 16x32): lanes 0-15 M=0..15 words0-7=K0..15,
// lanes 16-31 get +16 on K (same order as B).
// ---------------------------------------------------------------------------
__global__ __launch_bounds__(256) void vq_argmin(const float* __restrict__ z,
                                                 const unsigned* __restrict__ ws,
                                                 float* __restrict__ idx_out) {
  __shared__ unsigned sAhi[TOK_PER_CTA * 192];   // 96 KB
  __shared__ unsigned sAlo[TOK_PER_CTA * 192];   // 96 KB
  __shared__ unsigned long long part[TOK_PER_CTA];

  const int tid  = threadIdx.x;
  const int lane = tid & 31;
  const int wv   = tid >> 5;
  const int hf   = lane >> 4;
  const long tok0 = (long)blockIdx.x * TOK_PER_CTA;

  if (tid < TOK_PER_CTA) part[tid] = ~0ull;

  // Stage z tile as bf16 hi/lo A fragments. 128*192 pairs / 256 thr = 96 iters.
#pragma unroll 1
  for (int p = tid; p < TOK_PER_CTA * 192; p += 256) {
    const int mg = p / 192;
    const int t  = p - mg * 192;
    const int kc = t >> 4;
    const int r  = t & 15;
    const int h  = r >> 3;
    const int w  = r & 7;
    const int k  = kc * 32 + 16 * h + 2 * w;
    const float* zp = &z[(tok0 + mg) * EMB_DIM + k];
    unsigned hiw, low;
    split_pack(zp[0], zp[1], hiw, low);
    const int at = mg >> 4;
    const int m  = mg & 15;
    const int sl = 16 * h + m;
    const int off = ((at * KCHUNKS + kc) * 32 + sl) * 8 + w;
    sAhi[off] = hiw;
    sAlo[off] = low;
  }
  __syncthreads();

  const float* esq = (const float*)&ws[WS_ESQ];

#pragma unroll 1
  for (int ci = 0; ci < 4; ++ci) {
    const int ct = wv + 8 * ci;            // this wave's column tile (16 codes)
    v8f acc[ATILES];
#pragma unroll
    for (int at = 0; at < ATILES; ++at) { v8f zero = {}; acc[at] = zero; }

    // Double-buffered B fragments (L2-resident): prefetch kc+1 while the
    // 24 WMMAs of chunk kc execute. boff advances 32 lanes * 8 words per kc.
    const int bbase = (ct * KCHUNKS) * 32 * 8 + lane * 8;
    v16bf bhi = *(const v16bf*)&ws[WS_BHI + bbase];
    v16bf blo = *(const v16bf*)&ws[WS_BLO + bbase];

#pragma unroll 1
    for (int kc = 0; kc < KCHUNKS; ++kc) {
      const int kcn  = (kc + 1 < KCHUNKS) ? (kc + 1) : (KCHUNKS - 1);
      const int bnxt = bbase + kcn * 256;
      const v16bf bhiN = *(const v16bf*)&ws[WS_BHI + bnxt];
      const v16bf bloN = *(const v16bf*)&ws[WS_BLO + bnxt];

      const int abase = kc * 256 + lane * 8;   // within one atile slab
#pragma unroll
      for (int at = 0; at < ATILES; ++at) {
        const int aoff = at * (KCHUNKS * 256) + abase;
        const v16bf ahi = *(const v16bf*)&sAhi[aoff];
        const v16bf alo = *(const v16bf*)&sAlo[aoff];
        acc[at] = __builtin_amdgcn_wmma_f32_16x16x32_bf16(
            false, ahi, false, bhi, (short)0, acc[at], false, false);
        acc[at] = __builtin_amdgcn_wmma_f32_16x16x32_bf16(
            false, ahi, false, blo, (short)0, acc[at], false, false);
        acc[at] = __builtin_amdgcn_wmma_f32_16x16x32_bf16(
            false, alo, false, bhi, (short)0, acc[at], false, false);
      }
      bhi = bhiN;
      blo = bloN;
    }

    // distance = e_sq - 2*score (z^2 term constant per token: argmin-invariant)
    const int   code = ct * 16 + (lane & 15);
    const float eq   = esq[code];
#pragma unroll
    for (int at = 0; at < ATILES; ++at) {
#pragma unroll
      for (int r = 0; r < 8; ++r) {
        const float d = eq - 2.0f * acc[at][r];
        unsigned ub = __float_as_uint(d);
        ub = (ub & 0x80000000u) ? ~ub : (ub | 0x80000000u);  // ordered key
        unsigned long long v = ((unsigned long long)ub << 32) | (unsigned)code;
#pragma unroll
        for (int mk = 1; mk <= 8; mk <<= 1) {       // min within 16-lane half
          unsigned long long o = __shfl_xor(v, mk, 32);
          v = (o < v) ? o : v;
        }
        if ((lane & 15) == r)                        // token = at*16 + 8*hf + r
          atomicMin(&part[at * 16 + 8 * hf + r], v); // ds_min_u64
      }
    }
  }
  __syncthreads();
  if (tid < TOK_PER_CTA)
    idx_out[tok0 + tid] = (float)(unsigned)(part[tid] & 0xFFFFFFFFull);
}

// ---------------------------------------------------------------------------
// Kernel 3: gather z_q = z + (e - z) (reference straight-through FP order),
// accumulate (e - z)^2 with fixed-order block reduction.
// ---------------------------------------------------------------------------
__global__ __launch_bounds__(256) void vq_gather(const float* __restrict__ z,
                                                 const float* __restrict__ emb,
                                                 const float* __restrict__ idx_f,
                                                 float* __restrict__ zq,
                                                 float* __restrict__ part) {
  __shared__ float red[256];
  const int tid  = threadIdx.x;
  const int lane = tid & 31;
  const int wv   = tid >> 5;
  const long tok0 = (long)blockIdx.x * 128;
  float s = 0.f;
#pragma unroll 1
  for (int it = 0; it < 16; ++it) {
    const long tok = tok0 + it * 8 + wv;
    const int  idx = (int)idx_f[tok];
    const float* er = &emb[(long)idx * EMB_DIM];
    const float* zr = &z[tok * EMB_DIM];
    float*       qr = &zq[tok * EMB_DIM];
#pragma unroll
    for (int j = 0; j < KCHUNKS; ++j) {
      const int k = j * 32 + lane;
      const float e = er[k], zv = zr[k];
      const float t = e - zv;
      qr[k] = zv + t;                 // z + (z_q - z): matches reference fwd
      s += t * t;
    }
  }
  red[tid] = s;
  __syncthreads();
  for (int st = 128; st > 0; st >>= 1) {
    if (tid < st) red[tid] += red[tid + st];
    __syncthreads();
  }
  if (tid == 0) part[blockIdx.x] = red[0];
}

// ---------------------------------------------------------------------------
// Kernel 4: deterministic final loss reduction (2048 partials).
// ---------------------------------------------------------------------------
__global__ __launch_bounds__(256) void vq_loss(const float* __restrict__ part,
                                               float* __restrict__ out_loss) {
  __shared__ float red[256];
  const int tid = threadIdx.x;
  float s = 0.f;
  for (int i = 0; i < 8; ++i) s += part[tid * 8 + i];
  red[tid] = s;
  __syncthreads();
  for (int st = 128; st > 0; st >>= 1) {
    if (tid < st) red[tid] += red[tid + st];
    __syncthreads();
  }
  if (tid == 0) {
    const float mean = red[0] / (float)((long)N_TOKENS * EMB_DIM);
    *out_loss = mean * (1.0f + BETA);
  }
}

extern "C" void kernel_launch(void* const* d_in, const int* in_sizes, int n_in,
                              void* d_out, int out_size, void* d_ws, size_t ws_size,
                              hipStream_t stream) {
  const float* z   = (const float*)d_in[0];   // [N, 384] f32
  const float* emb = (const float*)d_in[1];   // [512, 384] f32
  float* out   = (float*)d_out;
  float* zq    = out;                                   // [N*384]
  float* idx_f = out + (size_t)N_TOKENS * EMB_DIM;      // [N]
  float* loss  = idx_f + N_TOKENS;                      // [1]
  unsigned* ws = (unsigned*)d_ws;
  float* part  = (float*)d_ws + WS_PART;                // [2048]

  vq_prep  <<<NUM_CODES, 192, 0, stream>>>(emb, ws);
  vq_argmin<<<N_TOKENS / TOK_PER_CTA, 256, 0, stream>>>(z, ws, idx_f);
  vq_gather<<<N_TOKENS / 128, 256, 0, stream>>>(z, emb, idx_f, zq, part);
  vq_loss  <<<1, 256, 0, stream>>>(part, loss);
}
